// QuantumSuperposition_27925877358854
// MI455X (gfx1250) — matrix operand order repaired and checked by reading, not compiled, not run on hardware
//
#include <hip/hip_runtime.h>
#include <hip/hip_bf16.h>
#include <math.h>

// ---------------------------------------------------------------------------
// Types for WMMA
// ---------------------------------------------------------------------------
typedef __bf16        v16bf __attribute__((ext_vector_type(16)));
typedef float         v8f   __attribute__((ext_vector_type(8)));
typedef unsigned int  v4u   __attribute__((ext_vector_type(4)));

union Frag { v4u u[2]; v16bf v; };

// ---------------------------------------------------------------------------
// bf16 helpers (round-to-nearest-even)
// ---------------------------------------------------------------------------
__device__ __forceinline__ unsigned short f2bf(float x) {
    unsigned u = __builtin_bit_cast(unsigned, x);
    u += 0x7FFFu + ((u >> 16) & 1u);
    return (unsigned short)(u >> 16);
}
__device__ __forceinline__ float bf2f(unsigned short h) {
    unsigned u = ((unsigned)h) << 16;
    return __builtin_bit_cast(float, u);
}

// ---------------------------------------------------------------------------
// Kernel 1: X f32 -> bf16 hi/lo split (row-major, contiguous)
// ---------------------------------------------------------------------------
__global__ __launch_bounds__(256)
void qsup_cvt_kernel(const float* __restrict__ src,
                     unsigned short* __restrict__ hi,
                     unsigned short* __restrict__ lo, int n) {
    int i = blockIdx.x * blockDim.x + threadIdx.x;
    int stride = gridDim.x * blockDim.x;
    for (; i < n; i += stride) {
        float x = src[i];
        unsigned short h = f2bf(x);
        float r = x - bf2f(h);
        hi[i] = h;
        lo[i] = f2bf(r);
    }
}

// ---------------------------------------------------------------------------
// Kernel 1b: W f32 -> bf16 hi/lo, chunk-transposed layout
//   WT[((s*32 + k/32)*1024 + q)*32 + k%32]
//   so each GEMM B tile (q0..q0+127 rows of 32 k-shorts) is one contiguous
//   8 KB block -> pure async DMA staging in the GEMM.
//   LDS 32x32 tile transpose keeps both global sides coalesced.
// ---------------------------------------------------------------------------
__global__ __launch_bounds__(256)
void qsup_cvtW_kernel(const float* __restrict__ W,
                      unsigned short* __restrict__ WTh,
                      unsigned short* __restrict__ WTl) {
    __shared__ float tile[32][33];
    const int b  = blockIdx.x;        // [s(4b) | kb(5b) | qb(5b)]
    const int qb = b & 31;
    const int kb = (b >> 5) & 31;
    const int s  = b >> 10;
    const int t  = threadIdx.x;

    const float* src = W + ((size_t)s << 20) + ((size_t)(kb * 32) << 10) + (size_t)(qb * 32);
#pragma unroll
    for (int i = t; i < 1024; i += 256) {
        int kk = i >> 5, qi = i & 31;               // lanes sweep qi -> coalesced
        tile[kk][qi] = src[kk * 1024 + qi];
    }
    __syncthreads();
    const size_t outBase = ((((size_t)(s * 32 + kb) << 10) + (size_t)(qb * 32)) << 5);
#pragma unroll
    for (int i = t; i < 1024; i += 256) {
        int qi = i >> 5, kk = i & 31;               // lanes sweep kk -> coalesced
        float x = tile[kk][qi];
        unsigned short h = f2bf(x);
        size_t o = outBase + ((size_t)qi << 5) + (size_t)kk;
        WTh[o] = h;
        WTl[o] = f2bf(x - bf2f(h));
    }
}

// ---------------------------------------------------------------------------
// Kernel 2: mix[s,q] = softmax_s(amp)[s,q] * cos(phase[s,q])
// ---------------------------------------------------------------------------
__global__ __launch_bounds__(256)
void qsup_mix_kernel(const float* __restrict__ amp,
                     const float* __restrict__ phase,
                     float* __restrict__ mix) {
    int q = blockIdx.x * blockDim.x + threadIdx.x;
    if (q >= 1024) return;
    float a[16];
    float m = -INFINITY;
#pragma unroll
    for (int s = 0; s < 16; ++s) { a[s] = amp[s * 1024 + q]; m = fmaxf(m, a[s]); }
    float sum = 0.0f;
#pragma unroll
    for (int s = 0; s < 16; ++s) { a[s] = expf(a[s] - m); sum += a[s]; }
    float inv = 1.0f / sum;
#pragma unroll
    for (int s = 0; s < 16; ++s)
        mix[s * 1024 + q] = a[s] * inv * cosf(phase[s * 1024 + q]);
}

// ---------------------------------------------------------------------------
// Kernel 3: bf16x3 WMMA GEMM, fully async-DMA staged, double-buffered LDS.
//   M = 4096 (b), N = 16384 (s*1024+q), K = 1024
//   Block: 256 threads (8 waves), tile 128x128, K-chunk 32, 2-stage ping-pong.
//   LDS: 2 x {Ah 8K, Al 8K, Bh 8K, Bl 8K} = 64 KB, unpadded 64 B rows with
//   16 B-block XOR swizzle (blk ^= (row>>2)&3) -> conflict-free ds_load_b128.
// ---------------------------------------------------------------------------
#define BM 128
#define BN 128
#define KC 32
#define K_DIM 1024
#define N_DIM 16384
#define STAGE_BYTES 32768   // AH | AL | BH | BL, 8 KB each

__device__ __forceinline__ void issue_stage(
    unsigned lds0,                       // LDS byte addr of this stage's base
    const unsigned short* XhBlk, const unsigned short* XlBlk,
    const unsigned short* WThBlk, const unsigned short* WTlBlk,
    int k0, int q0, int tid) {
    const int kb5 = (k0 >> 5) << 10;     // kb * 1024
#pragma unroll
    for (int j = 0; j < 2; ++j) {
        const int i = tid + j * 256;     // 512 x 16 B transfers per matrix pair
        // --- A: rows m of X (contiguous K) ---
        {
            int m = i >> 2, blk = i & 3;
            unsigned voff = ((unsigned)((m << 10) + k0) << 1) + ((unsigned)blk << 4);
            unsigned loff = ((unsigned)m << 6) + ((unsigned)((blk ^ (m >> 2)) & 3) << 4);
            asm volatile("global_load_async_to_lds_b128 %0, %1, %2"
                         :: "v"(lds0 + loff), "v"(voff), "s"(XhBlk) : "memory");
            asm volatile("global_load_async_to_lds_b128 %0, %1, %2"
                         :: "v"(lds0 + 8192u + loff), "v"(voff), "s"(XlBlk) : "memory");
        }
        // --- B: rows n of chunk-transposed W (contiguous K) ---
        {
            int n = i >> 2, blk = i & 3;
            unsigned voff = ((unsigned)(kb5 + q0 + n) << 6) + ((unsigned)blk << 4);
            unsigned loff = ((unsigned)n << 6) + ((unsigned)((blk ^ (n >> 2)) & 3) << 4);
            asm volatile("global_load_async_to_lds_b128 %0, %1, %2"
                         :: "v"(lds0 + 16384u + loff), "v"(voff), "s"(WThBlk) : "memory");
            asm volatile("global_load_async_to_lds_b128 %0, %1, %2"
                         :: "v"(lds0 + 24576u + loff), "v"(voff), "s"(WTlBlk) : "memory");
        }
    }
}

__device__ __forceinline__ void do_chunk(
    const char* sm, int boff,
    int waveM, int waveN, int half, int l15, v8f (&acc)[2][4]) {
    const char* pAh = sm + boff;
    const char* pAl = sm + boff + 8192;
    const char* pBh = sm + boff + 16384;
    const char* pBl = sm + boff + 24576;

    Frag a[2][2];   // [mi][0]=hi, [1]=lo
    Frag b[4][2];   // [ni][0]=hi, [1]=lo
#pragma unroll
    for (int mi = 0; mi < 2; ++mi) {
        int m  = waveM * 32 + mi * 16 + l15;
        int sw = (m >> 2) & 3;
        const char* r0 = pAh + m * 64;
        a[mi][0].u[0] = *(const v4u*)(r0 + ((half ^ sw) << 4));
        a[mi][0].u[1] = *(const v4u*)(r0 + (((2 + half) ^ sw) << 4));
        const char* r1 = pAl + m * 64;
        a[mi][1].u[0] = *(const v4u*)(r1 + ((half ^ sw) << 4));
        a[mi][1].u[1] = *(const v4u*)(r1 + (((2 + half) ^ sw) << 4));
    }
#pragma unroll
    for (int ni = 0; ni < 4; ++ni) {
        int n  = waveN * 64 + ni * 16 + l15;
        int sw = (n >> 2) & 3;
        const char* r0 = pBh + n * 64;
        b[ni][0].u[0] = *(const v4u*)(r0 + ((half ^ sw) << 4));
        b[ni][0].u[1] = *(const v4u*)(r0 + (((2 + half) ^ sw) << 4));
        const char* r1 = pBl + n * 64;
        b[ni][1].u[0] = *(const v4u*)(r1 + ((half ^ sw) << 4));
        b[ni][1].u[1] = *(const v4u*)(r1 + (((2 + half) ^ sw) << 4));
    }
#pragma unroll
    for (int mi = 0; mi < 2; ++mi)
#pragma unroll
        for (int ni = 0; ni < 4; ++ni) {
            acc[mi][ni] = __builtin_amdgcn_wmma_f32_16x16x32_bf16(
                false, a[mi][1].v, false, b[ni][0].v, (short)0, acc[mi][ni], false, false);
            acc[mi][ni] = __builtin_amdgcn_wmma_f32_16x16x32_bf16(
                false, a[mi][0].v, false, b[ni][1].v, (short)0, acc[mi][ni], false, false);
            acc[mi][ni] = __builtin_amdgcn_wmma_f32_16x16x32_bf16(
                false, a[mi][0].v, false, b[ni][0].v, (short)0, acc[mi][ni], false, false);
        }
}

__global__ __launch_bounds__(256)
void qsup_gemm_kernel(const unsigned short* __restrict__ Xh,
                      const unsigned short* __restrict__ Xl,
                      const unsigned short* __restrict__ WTh,
                      const unsigned short* __restrict__ WTl,
                      const float* __restrict__ bias,
                      const float* __restrict__ mix,
                      float* __restrict__ out) {
    __shared__ __align__(16) char smem[2 * STAGE_BYTES];   // 64 KB

    const int tid   = threadIdx.x;
    const int lane  = tid & 31;
    const int wave  = tid >> 5;
    const int waveM = wave >> 1;   // 0..3
    const int waveN = wave & 1;    // 0..1
    const int half  = lane >> 4;   // 0/1
    const int l15   = lane & 15;

    const int n0   = blockIdx.x * BN;
    const int row0 = blockIdx.y * BM;
    const int sIdx = n0 >> 10;
    const int q0   = n0 & 1023;

    // Uniform SGPR base pointers; per-thread offsets stay 32-bit.
    const unsigned short* XhBlk  = Xh  + (size_t)row0 * K_DIM;
    const unsigned short* XlBlk  = Xl  + (size_t)row0 * K_DIM;
    const unsigned short* WThBlk = WTh + ((size_t)sIdx << 20);
    const unsigned short* WTlBlk = WTl + ((size_t)sIdx << 20);

    const unsigned ldsBase = (unsigned)(size_t)&smem[0];   // ISA 10.2: addr[31:0]

    const v8f vzero = {0.f, 0.f, 0.f, 0.f, 0.f, 0.f, 0.f, 0.f};
    v8f acc[2][4];
#pragma unroll
    for (int mi = 0; mi < 2; ++mi)
#pragma unroll
        for (int ni = 0; ni < 4; ++ni) acc[mi][ni] = vzero;

    // ---- prologue: DMA chunk 0 into stage 0 ----
    issue_stage(ldsBase, XhBlk, XlBlk, WThBlk, WTlBlk, 0, q0, tid);
    asm volatile("s_wait_asynccnt 0" ::: "memory");
    __syncthreads();

    // ---- main loop: 2 chunks per iteration, ping-pong stages ----
    for (int k0 = 0; k0 < K_DIM; k0 += 2 * KC) {
        issue_stage(ldsBase + STAGE_BYTES, XhBlk, XlBlk, WThBlk, WTlBlk,
                    k0 + KC, q0, tid);
        do_chunk(smem, 0, waveM, waveN, half, l15, acc);
        asm volatile("s_wait_asynccnt 0" ::: "memory");
        __syncthreads();

        if (k0 + 2 * KC < K_DIM)
            issue_stage(ldsBase, XhBlk, XlBlk, WThBlk, WTlBlk,
                        k0 + 2 * KC, q0, tid);
        do_chunk(smem, STAGE_BYTES, waveM, waveN, half, l15, acc);
        if (k0 + 2 * KC < K_DIM) {
            asm volatile("s_wait_asynccnt 0" ::: "memory");
            __syncthreads();
        }
    }

    // ---- epilogue: sup = (acc + bias[n]) * mix[n] ----
#pragma unroll
    for (int ni = 0; ni < 4; ++ni) {
        int n = n0 + waveN * 64 + ni * 16 + l15;
        float mv = mix[n];
        float bv = bias[n];
#pragma unroll
        for (int mi = 0; mi < 2; ++mi) {
            int rbase = row0 + waveM * 32 + mi * 16 + half * 8;
            float* p = out + (size_t)rbase * N_DIM + (size_t)n;
#pragma unroll
            for (int r = 0; r < 8; ++r) {
                *p = (acc[mi][ni][r] + bv) * mv;
                p += N_DIM;
            }
        }
    }
}

// ---------------------------------------------------------------------------
// Kernel 4: LayerNorm over last dim (1024) per (b,s) row, in-place on d_out
// ---------------------------------------------------------------------------
__global__ __launch_bounds__(256)
void qsup_ln_kernel(float* __restrict__ out,
                    const float* __restrict__ gamma,
                    const float* __restrict__ beta) {
    __shared__ float s1[256];
    __shared__ float s2[256];
    const int t = threadIdx.x;
    float* row = out + (size_t)blockIdx.x * 1024;

    float4 x = reinterpret_cast<float4*>(row)[t];
    s1[t] = x.x + x.y + x.z + x.w;
    s2[t] = x.x * x.x + x.y * x.y + x.z * x.z + x.w * x.w;
    __syncthreads();
#pragma unroll
    for (int o = 128; o > 0; o >>= 1) {
        if (t < o) { s1[t] += s1[t + o]; s2[t] += s2[t + o]; }
        __syncthreads();
    }
    float mean = s1[0] * (1.0f / 1024.0f);
    float var  = s2[0] * (1.0f / 1024.0f) - mean * mean;
    float rs   = rsqrtf(var + 1e-5f);

    float4 g  = reinterpret_cast<const float4*>(gamma)[t];
    float4 be = reinterpret_cast<const float4*>(beta)[t];
    float4 y;
    y.x = (x.x - mean) * rs * g.x + be.x;
    y.y = (x.y - mean) * rs * g.y + be.y;
    y.z = (x.z - mean) * rs * g.z + be.z;
    y.w = (x.w - mean) * rs * g.w + be.w;
    reinterpret_cast<float4*>(row)[t] = y;
}

// ---------------------------------------------------------------------------
// Launch
// ---------------------------------------------------------------------------
extern "C" void kernel_launch(void* const* d_in, const int* in_sizes, int n_in,
                              void* d_out, int out_size, void* d_ws, size_t ws_size,
                              hipStream_t stream) {
    const float* X     = (const float*)d_in[0];  // [4096, 1024]
    const float* W     = (const float*)d_in[1];  // [16, 1024, 1024]
    const float* bias  = (const float*)d_in[2];  // [16, 1024]
    const float* amp   = (const float*)d_in[3];  // [16, 1024]
    const float* phase = (const float*)d_in[4];  // [16, 1024]
    const float* gamma = (const float*)d_in[5];  // [1024]
    const float* beta  = (const float*)d_in[6];  // [1024]
    float* out = (float*)d_out;                  // [4096, 16, 1024]

    // workspace layout (~80 MB)
    char* ws = (char*)d_ws;
    float* mix = (float*)ws;                                        // 64 KB
    unsigned short* Xh  = (unsigned short*)(ws + 64 * 1024);        // 8 MB
    unsigned short* Xl  = Xh  + (size_t)4096 * 1024;                // 8 MB
    unsigned short* WTh = Xl  + (size_t)4096 * 1024;                // 32 MB
    unsigned short* WTl = WTh + (size_t)16 * 1024 * 1024;           // 32 MB

    qsup_cvt_kernel<<<2048, 256, 0, stream>>>(X, Xh, Xl, 4096 * 1024);
    qsup_cvtW_kernel<<<16384, 256, 0, stream>>>(W, WTh, WTl);
    qsup_mix_kernel<<<4, 256, 0, stream>>>(amp, phase, mix);

    dim3 grid(N_DIM / BN, 4096 / BM);  // 128 x 32
    qsup_gemm_kernel<<<grid, 256, 0, stream>>>(Xh, Xl, WTh, WTl, bias, mix, out);

    qsup_ln_kernel<<<65536, 256, 0, stream>>>(out, gamma, beta);
}
